// PerformerViTTactile_52673478918827
// MI455X (gfx1250) — matrix operand order repaired and checked by reading, not compiled, not run on hardware
//
#include <hip/hip_runtime.h>
#include <math.h>

// ---------------------------------------------------------------------------
// Types / helpers
// ---------------------------------------------------------------------------
typedef __attribute__((ext_vector_type(16))) __bf16 v16bf;
typedef __attribute__((ext_vector_type(8)))  float  v8f;

#define DEV static __device__ __forceinline__

DEV unsigned short f2bf_bits(float f) {
  unsigned u = __builtin_bit_cast(unsigned, f);
  unsigned r = 0x7FFFu + ((u >> 16) & 1u);
  u += r;
  return (unsigned short)(u >> 16);
}
DEV __bf16 f2bf(float f) {
  unsigned short b = f2bf_bits(f);
  return __builtin_bit_cast(__bf16, b);
}
DEV float bf2f(__bf16 h) {
  unsigned short b = __builtin_bit_cast(unsigned short, h);
  unsigned u = ((unsigned)b) << 16;
  return __builtin_bit_cast(float, u);
}
DEV __bf16 bf0() {
  unsigned short z = 0;
  return __builtin_bit_cast(__bf16, z);
}
DEV v8f zero8() { v8f z; for (int i = 0; i < 8; ++i) z[i] = 0.f; return z; }

// Generic pointer -> LDS byte offset (aperture map: addr[31:0] is the
// wave-relative LDS offset for shared-segment generic addresses).
DEV unsigned lds_off(const void* p) { return (unsigned)(unsigned long long)p; }

// CDNA5 async global->LDS copy (ASYNCcnt-tracked, bypasses VGPRs).
DEV void async_b128(unsigned ldsoff, const void* gaddr) {
  asm volatile("global_load_async_to_lds_b128 %0, %1, off"
               :: "v"(ldsoff), "v"(gaddr) : "memory");
}
DEV void wait_async_le3() { asm volatile("s_wait_asynccnt 0x3" ::: "memory"); }
DEV void wait_async_0()   { asm volatile("s_wait_asynccnt 0x0" ::: "memory"); }

// Architecture constants
#define NB    64          // batch
#define LTOK  197         // tokens (196 patches + cls)
#define NROW  (NB*LTOK)   // 12608
#define DMODEL 192
#define NHEAD 3
#define DHEAD 64
#define MFEAT 128         // random features (phi dim = 2M = 256)
#define NDEPTH 12
#define TEMP_F   0.125f           // 1/sqrt(64)
#define SQRT_TEMP 0.35355339059327373f
#define PHI_SCALE 0.0625f         // 1/sqrt(2M) = 1/16

// ---------------------------------------------------------------------------
// Elementwise conversion / packing kernels
// ---------------------------------------------------------------------------
__global__ void k_f32_to_bf16(const float* __restrict__ src, __bf16* __restrict__ dst, int n) {
  int i = blockIdx.x * blockDim.x + threadIdx.x;
  if (i < n) dst[i] = f2bf(src[i]);
}

__global__ void k_pack_qkv_w(const float* __restrict__ Wq, const float* __restrict__ Wk,
                             const float* __restrict__ Wv, __bf16* __restrict__ dst) {
  int i = blockIdx.x * blockDim.x + threadIdx.x;   // 12*576*192
  if (i >= NDEPTH * 576 * DMODEL) return;
  int l = i / (576 * DMODEL);
  int rem = i % (576 * DMODEL);
  int r = rem / DMODEL, c = rem % DMODEL;
  const float* s = (r < 192) ? Wq : (r < 384) ? Wk : Wv;
  dst[i] = f2bf(s[((size_t)l * DMODEL + (r % DMODEL)) * DMODEL + c]);
}

__global__ void k_pack_qkv_b(const float* __restrict__ bq, const float* __restrict__ bk,
                             const float* __restrict__ bv, float* __restrict__ dst) {
  int i = blockIdx.x * blockDim.x + threadIdx.x;   // 12*576
  if (i >= NDEPTH * 576) return;
  int l = i / 576, r = i % 576;
  const float* s = (r < 192) ? bq : (r < 384) ? bk : bv;
  dst[i] = s[l * DMODEL + (r % DMODEL)];
}

__global__ void k_omegaT(const float* __restrict__ omega, __bf16* __restrict__ dst) {
  // omega [12][64][128] -> omegaT [12][128][64]  ([N][K] layout for WMMA B)
  int i = blockIdx.x * blockDim.x + threadIdx.x;   // 12*128*64
  if (i >= NDEPTH * MFEAT * DHEAD) return;
  int l = i / (MFEAT * DHEAD);
  int rem = i % (MFEAT * DHEAD);
  int m = rem / DHEAD, d = rem % DHEAD;
  dst[i] = f2bf(omega[((size_t)l * DHEAD + d) * MFEAT + m]);
}

__global__ void k_im2col(const float* __restrict__ x, __bf16* __restrict__ Ap) {
  // Ap[b*196 + h*14 + w][c*256 + p*16 + q] = x[b][c][h*16+p][w*16+q]
  int i = blockIdx.x * blockDim.x + threadIdx.x;   // 12544*768
  if (i >= 12544 * 768) return;
  int m = i / 768, k = i % 768;
  int b = m / 196, t = m % 196, hh = t / 14, ww = t % 14;
  int c = k / 256, rr = (k % 256) / 16, cc = k % 16;
  Ap[i] = f2bf(x[(((size_t)b * 3 + c) * 224 + hh * 16 + rr) * 224 + ww * 16 + cc]);
}

// ---------------------------------------------------------------------------
// Generic bf16 WMMA GEMM:  C[M,N] = A[M,K] (row-major, lda) x W[N,K]^T (ldw)
// Block tile 128x64, 8 waves, wave tile 32x32 (2x2 WMMA 16x16x32 tiles).
// Double-buffered LDS filled by GLOBAL_LOAD_ASYNC_TO_LDS_B128 (ASYNCcnt).
// OOB tile rows/cols are clamped (not zeroed): C[m,n] depends only on A row m
// and W row n, and OOB m/n are masked at the epilogue stores.
// flags: 1=bias, 2=relu, 4=residual add (f32, ldr)
// ---------------------------------------------------------------------------
#define LDA_P 40   // padded LDS row stride (bf16 elems) to spread banks
#define ASZ (128 * LDA_P)
#define BSZ (64 * LDA_P)

__global__ __launch_bounds__(256) void k_gemm(
    const __bf16* __restrict__ A, int lda,
    const __bf16* __restrict__ W, int ldw,
    float* __restrict__ outF, __bf16* __restrict__ outB, int ldc,
    const float* __restrict__ bias,
    const float* __restrict__ resid, int ldr,
    int M, int N, int K, int flags) {
  __shared__ __align__(16) __bf16 lA[2 * ASZ];
  __shared__ __align__(16) __bf16 lB[2 * BSZ];

  const int tid = threadIdx.x;
  const int lane = tid & 31;
  const int w = tid >> 5;
  const int m0 = blockIdx.x * 128;
  const int n0 = blockIdx.y * 64;
  const int wm = w & 3, wn = w >> 2;      // 4x2 wave grid
  const int ml = lane & 15, kg = lane >> 4;

  v8f acc[2][2];
  for (int i = 0; i < 2; ++i)
    for (int j = 0; j < 2; ++j) acc[i][j] = zero8();

  // loader mapping (all 256 threads, uniform 3 async B128s per tile)
  const int ar = tid >> 1;              // A row 0..127
  const int ah = (tid & 1) * 16;        // A col half (bf16 elems)
  const int br = tid >> 2;              // W row 0..63
  const int bq = (tid & 3) * 8;         // W col quarter (bf16 elems)
  int gmA = m0 + ar; if (gmA > M - 1) gmA = M - 1;   // clamp (safe, masked at store)
  int gnB = n0 + br; if (gnB > N - 1) gnB = N - 1;
  const __bf16* gA = A + (size_t)gmA * lda + ah;
  const __bf16* gW = W + (size_t)gnB * ldw + bq;
  const unsigned laA = lds_off(&lA[ar * LDA_P + ah]);
  const unsigned laB = lds_off(&lB[br * LDA_P + bq]);

  // prologue: tile 0 -> buffer 0
  async_b128(laA,      gA);
  async_b128(laA + 16, (const char*)gA + 16);
  async_b128(laB,      gW);

  int buf = 0;
  for (int k0 = 0; k0 < K; k0 += 32) {
    const bool hasNext = (k0 + 32) < K;
    if (hasNext) {
      int nb = buf ^ 1;
      const __bf16* pa = gA + k0 + 32;
      const __bf16* pw = gW + k0 + 32;
      unsigned oa = laA + nb * (ASZ * 2);
      unsigned ob = laB + nb * (BSZ * 2);
      async_b128(oa,      pa);
      async_b128(oa + 16, (const char*)pa + 16);
      async_b128(ob,      pw);
      wait_async_le3();   // tile k0's 3 loads complete (in-order ASYNCcnt)
    } else {
      wait_async_0();
    }
    __syncthreads();      // tile k0 visible from every wave

    const __bf16* bA = &lA[buf * ASZ];
    const __bf16* bB = &lB[buf * BSZ];
    v16bf afr[2], bfr[2];
    for (int mt = 0; mt < 2; ++mt) {
      const __bf16* pr = &bA[(wm * 32 + mt * 16 + ml) * LDA_P];
      for (int j = 0; j < 16; ++j) afr[mt][j] = pr[kg * 16 + j];        // K 0..15 / 16..31
    }
    for (int nt = 0; nt < 2; ++nt) {
      const __bf16* pr = &bB[(wn * 32 + nt * 16 + ml) * LDA_P + kg * 16];
      for (int j = 0; j < 16; ++j) bfr[nt][j] = pr[j];                  // K 0..15 / 16..31
    }
    for (int mt = 0; mt < 2; ++mt)
      for (int nt = 0; nt < 2; ++nt)
        acc[mt][nt] = __builtin_amdgcn_wmma_f32_16x16x32_bf16(
            false, afr[mt], false, bfr[nt], (short)0, acc[mt][nt], false, false);
    __syncthreads();      // everyone done reading buf before it is refilled
    buf ^= 1;
  }

  // ---- epilogue: C layout: lane -> N = lane&15, M = r + 8*(lane>>4) ----
  for (int mt = 0; mt < 2; ++mt)
    for (int nt = 0; nt < 2; ++nt) {
      int n = n0 + wn * 32 + nt * 16 + ml;
      int mBase = m0 + wm * 32 + mt * 16 + kg * 8;
      if (n < N) {
        float bi = (flags & 1) ? bias[n] : 0.f;
        for (int r = 0; r < 8; ++r) {
          int m = mBase + r;
          if (m < M) {
            float vv = acc[mt][nt][r] + bi;
            if (flags & 2) vv = vv > 0.f ? vv : 0.f;
            if (flags & 4) vv += resid[(size_t)m * ldr + n];
            if (outF) outF[(size_t)m * ldc + n] = vv;
            if (outB) outB[(size_t)m * ldc + n] = f2bf(vv);
          }
        }
      }
    }
}

// ---------------------------------------------------------------------------
// Batched Performer core:  KV^T[b,h][d][feat] = (Kp^T V)^T,  K = L = 197
// One block per (b,h). 8 waves: wave w owns feature rows [32w, 32w+32), all 64 d.
// ---------------------------------------------------------------------------
__global__ __launch_bounds__(256) void k_kv(
    const __bf16* __restrict__ Kp,   // [NROW][3][256]
    const __bf16* __restrict__ qkv,  // [NROW][576], v at col 384+h*64
    __bf16* __restrict__ KVt) {      // [192][64][256]
  int batch = blockIdx.x;
  int b = batch / NHEAD, h = batch % NHEAD;
  int tid = threadIdx.x, lane = tid & 31, w = tid >> 5;
  int ml = lane & 15, kg = lane >> 4;

  v8f acc[2][4];
  for (int i = 0; i < 2; ++i)
    for (int j = 0; j < 4; ++j) acc[i][j] = zero8();

  for (int s0 = 0; s0 < LTOK; s0 += 32) {
    v16bf afr[2], bfr[4];
    for (int mt = 0; mt < 2; ++mt) {
      int mf = w * 32 + mt * 16 + ml;   // feature row (A = Kp^T: gather along s)
      for (int j = 0; j < 16; ++j) {
        int kk = kg * 16 + j;
        int s = s0 + kk;
        afr[mt][j] = (s < LTOK) ? Kp[((size_t)(b * LTOK + s)) * 768 + h * 256 + mf] : bf0();
      }
    }
    for (int nt = 0; nt < 4; ++nt) {
      int d = nt * 16 + ml;             // B = V: column d, gather along s
      for (int j = 0; j < 16; ++j) {
        int s = s0 + kg * 16 + j;
        bfr[nt][j] = (s < LTOK) ? qkv[((size_t)(b * LTOK + s)) * 576 + 384 + h * 64 + d] : bf0();
      }
    }
    for (int mt = 0; mt < 2; ++mt)
      for (int nt = 0; nt < 4; ++nt)
        acc[mt][nt] = __builtin_amdgcn_wmma_f32_16x16x32_bf16(
            false, afr[mt], false, bfr[nt], (short)0, acc[mt][nt], false, false);
  }
  // store transposed: KVt[batch][d][feat] (so next GEMM's B frags are contiguous)
  for (int mt = 0; mt < 2; ++mt)
    for (int nt = 0; nt < 4; ++nt) {
      int d = nt * 16 + ml;
      int mfB = w * 32 + mt * 16 + kg * 8;
      for (int r = 0; r < 8; ++r)
        KVt[((size_t)batch * 64 + d) * 256 + mfB + r] = f2bf(acc[mt][nt][r]);
    }
}

// attn[b,l,h*64+d] = Z[b,l,h] * Qp[b,l,h,:] . KV[:, d]   (K = 256)
__global__ __launch_bounds__(256) void k_attn(
    const __bf16* __restrict__ Qp,   // [NROW][3][256]
    const __bf16* __restrict__ KVt,  // [192][64][256]
    const float* __restrict__ Z,     // [NROW][3]
    __bf16* __restrict__ attn) {     // [NROW][192]
  int batch = blockIdx.x;
  int b = batch / NHEAD, h = batch % NHEAD;
  int tid = threadIdx.x, lane = tid & 31, w = tid >> 5;
  int ml = lane & 15, kg = lane >> 4;

  v8f acc[2][4];
  for (int i = 0; i < 2; ++i)
    for (int j = 0; j < 4; ++j) acc[i][j] = zero8();

  for (int k0 = 0; k0 < 256; k0 += 32) {
    v16bf afr[2], bfr[4];
    for (int u = 0; u < 2; ++u) {
      int mt = w + u * 8;              // 13 valid m-tiles (197 rows)
      int l = mt * 16 + ml;
      for (int j = 0; j < 16; ++j) {
        int kk = kg * 16 + j;
        afr[u][j] = (mt < 13 && l < LTOK)
                        ? Qp[((size_t)(b * LTOK + l)) * 768 + h * 256 + k0 + kk]
                        : bf0();
      }
    }
    for (int nt = 0; nt < 4; ++nt) {
      int d = nt * 16 + ml;
      const __bf16* p = &KVt[((size_t)batch * 64 + d) * 256 + k0 + kg * 16];
      for (int j = 0; j < 16; ++j) bfr[nt][j] = p[j];
    }
    for (int u = 0; u < 2; ++u)
      for (int nt = 0; nt < 4; ++nt)
        acc[u][nt] = __builtin_amdgcn_wmma_f32_16x16x32_bf16(
            false, afr[u], false, bfr[nt], (short)0, acc[u][nt], false, false);
  }
  for (int u = 0; u < 2; ++u) {
    int mt = w + u * 8;
    if (mt >= 13) continue;
    for (int nt = 0; nt < 4; ++nt) {
      int d = nt * 16 + ml;
      int lBase = mt * 16 + kg * 8;
      for (int r = 0; r < 8; ++r) {
        int l = lBase + r;
        if (l < LTOK) {
          float z = Z[((size_t)(b * LTOK + l)) * NHEAD + h];
          attn[((size_t)(b * LTOK + l)) * DMODEL + h * 64 + d] = f2bf(acc[u][nt][r] * z);
        }
      }
    }
  }
}

// ---------------------------------------------------------------------------
// FAVOR+ phi:  phi = exp(+-s*u - 0.5*TEMP*|q|^2) / 16    (128 threads / row)
// ---------------------------------------------------------------------------
__global__ void k_phi(const float* __restrict__ U,      // [NROW][128]
                      const __bf16* __restrict__ qkv,   // [NROW][576]
                      int colbase, int h, __bf16* __restrict__ dst /* Qp or Kp */) {
  int row = blockIdx.x;
  int t = threadIdx.x;  // 128
  __shared__ float red[64];
  __shared__ float s_off;
  if (t < 64) {
    float v = bf2f(qkv[(size_t)row * 576 + colbase + t]);
    red[t] = v * v;
  }
  __syncthreads();
  for (int off = 32; off > 0; off >>= 1) {
    if (t < off && t + off < 64) red[t] += red[t + off];
    __syncthreads();
  }
  if (t == 0) s_off = 0.5f * TEMP_F * red[0];
  __syncthreads();
  float u = U[(size_t)row * MFEAT + t] * SQRT_TEMP;
  float off = s_off;
  dst[(size_t)row * 768 + h * 256 + t]       = f2bf(expf(u - off) * PHI_SCALE);
  dst[(size_t)row * 768 + h * 256 + 128 + t] = f2bf(expf(-u - off) * PHI_SCALE);
}

__global__ void k_ksum(const __bf16* __restrict__ Kp, float* __restrict__ Ksum) {
  int g = blockIdx.x * blockDim.x + threadIdx.x;  // 64*768
  if (g >= NB * 768) return;
  int b = g / 768, rem = g % 768;
  float s = 0.f;
  for (int l = 0; l < LTOK; ++l) s += bf2f(Kp[((size_t)(b * LTOK + l)) * 768 + rem]);
  Ksum[g] = s;
}

__global__ void k_z(const __bf16* __restrict__ Qp, const float* __restrict__ Ksum,
                    float* __restrict__ Z) {
  int g = blockIdx.x * blockDim.x + threadIdx.x;  // NROW*3
  if (g >= NROW * NHEAD) return;
  int row = g / NHEAD, h = g % NHEAD, b = row / LTOK;
  const __bf16* q = &Qp[(size_t)row * 768 + h * 256];
  const float* ks = &Ksum[(size_t)b * 768 + h * 256];
  float s = 0.f;
  for (int j = 0; j < 256; ++j) s += bf2f(q[j]) * ks[j];
  Z[g] = 1.f / (s + 1e-6f);
}

// ---------------------------------------------------------------------------
// LayerNorm over 192 (64 threads/row, 3 cols each); optional second LN fused.
// ---------------------------------------------------------------------------
__global__ void k_ln(const float* __restrict__ in,
                     const float* __restrict__ g1, const float* __restrict__ b1,
                     const float* __restrict__ g2, const float* __restrict__ b2,
                     float* __restrict__ outF, __bf16* __restrict__ outB, int twoStage) {
  int row = blockIdx.x;
  int t = threadIdx.x;  // 64
  __shared__ float s1[64], s2[64];
  float v[3];
  for (int j = 0; j < 3; ++j) v[j] = in[(size_t)row * DMODEL + t * 3 + j];
  float ls = 0.f, lq = 0.f;
  for (int j = 0; j < 3; ++j) { ls += v[j]; lq += v[j] * v[j]; }
  s1[t] = ls; s2[t] = lq;
  __syncthreads();
  for (int off = 32; off > 0; off >>= 1) {
    if (t < off) { s1[t] += s1[t + off]; s2[t] += s2[t + off]; }
    __syncthreads();
  }
  float mu = s1[0] * (1.f / DMODEL);
  float var = s2[0] * (1.f / DMODEL) - mu * mu;
  float rs = rsqrtf(var + 1e-5f);
  for (int j = 0; j < 3; ++j) v[j] = (v[j] - mu) * rs * g1[t * 3 + j] + b1[t * 3 + j];
  if (twoStage) {
    __syncthreads();
    ls = 0.f; lq = 0.f;
    for (int j = 0; j < 3; ++j) { ls += v[j]; lq += v[j] * v[j]; }
    s1[t] = ls; s2[t] = lq;
    __syncthreads();
    for (int off = 32; off > 0; off >>= 1) {
      if (t < off) { s1[t] += s1[t + off]; s2[t] += s2[t + off]; }
      __syncthreads();
    }
    mu = s1[0] * (1.f / DMODEL);
    var = s2[0] * (1.f / DMODEL) - mu * mu;
    rs = rsqrtf(var + 1e-5f);
    for (int j = 0; j < 3; ++j) v[j] = (v[j] - mu) * rs * g2[t * 3 + j] + b2[t * 3 + j];
  }
  for (int j = 0; j < 3; ++j) {
    if (outF) outF[(size_t)row * DMODEL + t * 3 + j] = v[j];
    if (outB) outB[(size_t)row * DMODEL + t * 3 + j] = f2bf(v[j]);
  }
}

__global__ void k_assemble(const float* __restrict__ lnout, const float* __restrict__ cls,
                           const float* __restrict__ pos, float* __restrict__ xc,
                           __bf16* __restrict__ xcb) {
  int g = blockIdx.x * blockDim.x + threadIdx.x;  // NB*197*192
  if (g >= NROW * DMODEL) return;
  int d = g % DMODEL;
  int rem = g / DMODEL;
  int l = rem % LTOK, b = rem / LTOK;
  float v;
  if (l == 0) v = cls[d] + pos[d];
  else v = lnout[((size_t)(b * 196 + l - 1)) * DMODEL + d] + pos[l * DMODEL + d];
  xc[g] = v;
  xcb[g] = f2bf(v);
}

__global__ void k_mean(const float* __restrict__ xc, float* __restrict__ zm,
                       __bf16* __restrict__ zb) {
  int g = blockIdx.x * blockDim.x + threadIdx.x;  // 64*192
  if (g >= NB * DMODEL) return;
  int b = g / DMODEL, d = g % DMODEL;
  float s = 0.f;
  for (int l = 0; l < LTOK; ++l) s += xc[((size_t)(b * LTOK + l)) * DMODEL + d];
  float v = s * (1.f / LTOK);
  zm[g] = v;
  zb[g] = f2bf(v);
}

// ---------------------------------------------------------------------------
// Host orchestration
// ---------------------------------------------------------------------------
static inline void gemm(hipStream_t s, const __bf16* A, int lda, const __bf16* W, int ldw,
                        float* oF, __bf16* oB, int ldc, const float* bias,
                        const float* res, int ldr, int M, int N, int K, int flags) {
  dim3 g((M + 127) / 128, (N + 63) / 64);
  k_gemm<<<g, 256, 0, s>>>(A, lda, W, ldw, oF, oB, ldc, bias, res, ldr, M, N, K, flags);
}

extern "C" void kernel_launch(void* const* d_in, const int* in_sizes, int n_in,
                              void* d_out, int out_size, void* d_ws, size_t ws_size,
                              hipStream_t stream) {
  const float* x        = (const float*)d_in[0];
  const float* patch_w  = (const float*)d_in[1];
  const float* patch_b  = (const float*)d_in[2];
  const float* pe_ln_g  = (const float*)d_in[3];
  const float* pe_ln_b  = (const float*)d_in[4];
  const float* cls      = (const float*)d_in[5];
  const float* pos      = (const float*)d_in[6];
  const float* Wq       = (const float*)d_in[7];
  const float* bq       = (const float*)d_in[8];
  const float* Wk       = (const float*)d_in[9];
  const float* bk       = (const float*)d_in[10];
  const float* Wv       = (const float*)d_in[11];
  const float* bv       = (const float*)d_in[12];
  const float* Wo       = (const float*)d_in[13];
  const float* bo       = (const float*)d_in[14];
  const float* ln1_g    = (const float*)d_in[15];
  const float* ln1_b    = (const float*)d_in[16];
  const float* ln2_g    = (const float*)d_in[17];
  const float* ln2_b    = (const float*)d_in[18];
  const float* lnb_g    = (const float*)d_in[19];
  const float* lnb_b    = (const float*)d_in[20];
  const float* W1       = (const float*)d_in[21];
  const float* b1       = (const float*)d_in[22];
  const float* W2       = (const float*)d_in[23];
  const float* b2       = (const float*)d_in[24];
  const float* omega    = (const float*)d_in[25];
  const float* head_w   = (const float*)d_in[26];
  const float* head_b   = (const float*)d_in[27];
  float* out = (float*)d_out;
  (void)in_sizes; (void)n_in; (void)out_size; (void)ws_size;

  // ---- workspace arena ----
  char* ws = (char*)d_ws;
  size_t off = 0;
  auto alloc = [&](size_t bytes) -> char* {
    char* p = ws + off;
    off += (bytes + 255) & ~(size_t)255;
    return p;
  };
  __bf16* wqkvB = (__bf16*)alloc((size_t)NDEPTH * 576 * 192 * 2);
  __bf16* woB   = (__bf16*)alloc((size_t)NDEPTH * 192 * 192 * 2);
  __bf16* w1B   = (__bf16*)alloc((size_t)NDEPTH * 768 * 192 * 2);
  __bf16* w2B   = (__bf16*)alloc((size_t)NDEPTH * 192 * 768 * 2);
  __bf16* wpB   = (__bf16*)alloc((size_t)192 * 768 * 2);
  __bf16* whB   = (__bf16*)alloc((size_t)3 * 1000 * 192 * 2);
  __bf16* omT   = (__bf16*)alloc((size_t)NDEPTH * 128 * 64 * 2);
  float*  bqkv  = (float*)alloc((size_t)NDEPTH * 576 * 4);
  float*  xc    = (float*)alloc((size_t)NROW * DMODEL * 4);
  __bf16* xcb   = (__bf16*)alloc((size_t)NROW * DMODEL * 2);
  float*  preLN = (float*)alloc((size_t)NROW * DMODEL * 4);   // also patch pre-LN
  __bf16* qkvB  = (__bf16*)alloc((size_t)NROW * 576 * 2);
  float*  U     = (float*)alloc((size_t)NROW * 128 * 4);
  __bf16* Qp    = (__bf16*)alloc((size_t)NROW * 768 * 2);
  __bf16* Kp    = (__bf16*)alloc((size_t)NROW * 768 * 2);
  float*  Ksum  = (float*)alloc((size_t)NB * 768 * 4);
  __bf16* KVt   = (__bf16*)alloc((size_t)NB * NHEAD * 64 * 256 * 2);
  float*  Zbuf  = (float*)alloc((size_t)NROW * NHEAD * 4);
  __bf16* attnB = (__bf16*)alloc((size_t)NROW * DMODEL * 2);
  __bf16* bigB  = (__bf16*)alloc((size_t)NROW * 768 * 2);     // Ap (im2col) then FFN hidden
  float*  zm    = (float*)alloc((size_t)NB * DMODEL * 4);
  __bf16* zb    = (__bf16*)alloc((size_t)NB * DMODEL * 2);

  auto cvt = [&](const float* s, __bf16* d, int n) {
    k_f32_to_bf16<<<(n + 255) / 256, 256, 0, stream>>>(s, d, n);
  };

  // ---- one-time conversions ----
  k_pack_qkv_w<<<(NDEPTH * 576 * 192 + 255) / 256, 256, 0, stream>>>(Wq, Wk, Wv, wqkvB);
  k_pack_qkv_b<<<(NDEPTH * 576 + 255) / 256, 256, 0, stream>>>(bq, bk, bv, bqkv);
  k_omegaT<<<(NDEPTH * 128 * 64 + 255) / 256, 256, 0, stream>>>(omega, omT);
  cvt(Wo, woB, NDEPTH * 192 * 192);
  cvt(W1, w1B, NDEPTH * 768 * 192);
  cvt(W2, w2B, NDEPTH * 192 * 768);
  cvt(patch_w, wpB, 192 * 768);
  cvt(head_w, whB, 3 * 1000 * 192);

  // ---- patch embed: im2col -> GEMM -> LN -> assemble tokens ----
  k_im2col<<<(12544 * 768 + 255) / 256, 256, 0, stream>>>(x, bigB);
  gemm(stream, bigB, 768, wpB, 768, preLN, nullptr, DMODEL, patch_b, nullptr, 0,
       12544, DMODEL, 768, /*bias*/1);
  k_ln<<<12544, 64, 0, stream>>>(preLN, pe_ln_g, pe_ln_b, nullptr, nullptr,
                                 preLN, nullptr, 0);
  k_assemble<<<(NROW * DMODEL + 255) / 256, 256, 0, stream>>>(preLN, cls, pos, xc, xcb);

  // ---- transformer layers ----
  for (int l = 0; l < NDEPTH; ++l) {
    // fused QKV projection
    gemm(stream, xcb, DMODEL, wqkvB + (size_t)l * 576 * 192, DMODEL,
         nullptr, qkvB, 576, bqkv + (size_t)l * 576, nullptr, 0,
         NROW, 576, DMODEL, 1);
    // FAVOR features: u = q@omega per (head, q/k), then phi
    for (int h = 0; h < NHEAD; ++h)
      for (int qk = 0; qk < 2; ++qk) {
        int colbase = qk * 192 + h * 64;
        gemm(stream, qkvB + colbase, 576, omT + (size_t)l * 128 * 64, 64,
             U, nullptr, 128, nullptr, nullptr, 0, NROW, 128, 64, 0);
        k_phi<<<NROW, 128, 0, stream>>>(U, qkvB, colbase, h, qk ? Kp : Qp);
      }
    k_ksum<<<(NB * 768 + 255) / 256, 256, 0, stream>>>(Kp, Ksum);
    k_z<<<(NROW * NHEAD + 255) / 256, 256, 0, stream>>>(Qp, Ksum, Zbuf);
    k_kv<<<NB * NHEAD, 256, 0, stream>>>(Kp, qkvB, KVt);
    k_attn<<<NB * NHEAD, 256, 0, stream>>>(Qp, KVt, Zbuf, attnB);
    // output projection + residual
    gemm(stream, attnB, DMODEL, woB + (size_t)l * 192 * 192, DMODEL,
         preLN, nullptr, DMODEL, bo + (size_t)l * DMODEL, xc, DMODEL,
         NROW, DMODEL, DMODEL, 1 | 4);
    k_ln<<<NROW, 64, 0, stream>>>(preLN, ln1_g + (size_t)l * DMODEL, ln1_b + (size_t)l * DMODEL,
                                  nullptr, nullptr, xc, xcb, 0);
    // FFN
    gemm(stream, xcb, DMODEL, w1B + (size_t)l * 768 * 192, DMODEL,
         nullptr, bigB, 768, b1 + (size_t)l * 768, nullptr, 0,
         NROW, 768, DMODEL, 1 | 2);
    gemm(stream, bigB, 768, w2B + (size_t)l * 192 * 768, 768,
         preLN, nullptr, DMODEL, b2 + (size_t)l * DMODEL, xc, DMODEL,
         NROW, DMODEL, 768, 1 | 4);
    // ln2 then per-block final norm (fused double LN)
    k_ln<<<NROW, 64, 0, stream>>>(preLN, ln2_g + (size_t)l * DMODEL, ln2_b + (size_t)l * DMODEL,
                                  lnb_g + (size_t)l * DMODEL, lnb_b + (size_t)l * DMODEL,
                                  xc, xcb, 1);
    // early exits
    if (l == 3 || l == 7 || l == 11) {
      int j = (l == 3) ? 0 : (l == 7) ? 1 : 2;
      k_mean<<<(NB * DMODEL + 255) / 256, 256, 0, stream>>>(xc, zm, zb);
      gemm(stream, zb, DMODEL, whB + (size_t)j * 1000 * 192, DMODEL,
           out + (size_t)j * NB * 1000, nullptr, 1000,
           head_b + (size_t)j * 1000, nullptr, 0, NB, 1000, DMODEL, 1);
    }
  }
}
